// MockMoEModel_12292196401256
// MI455X (gfx1250) — matrix operand order repaired and checked by reading, not compiled, not run on hardware
//
#include <hip/hip_runtime.h>
#include <hip/hip_bf16.h>

typedef __attribute__((ext_vector_type(16))) __bf16 v16bf;
typedef __attribute__((ext_vector_type(8)))  __bf16 v8bf;
typedef __attribute__((ext_vector_type(8)))  float  v8f;

static constexpr int H = 1024;     // hidden dim
static constexpr int T = 16384;    // B*S tokens
static constexpr int E = 8;        // experts
static constexpr int LAYERS = 2;

__device__ __forceinline__ void atomic_add_f32(float* p, float v) {
    __hip_atomic_fetch_add(p, v, __ATOMIC_RELAXED, __HIP_MEMORY_SCOPE_AGENT);
}

// ---------------------------------------------------------------------------
// fp32 -> bf16 bulk convert (grid-stride, 8 elements / thread / step)
// ---------------------------------------------------------------------------
__global__ __launch_bounds__(256) void cvt_f32_bf16_kernel(
    const float* __restrict__ src, __bf16* __restrict__ dst, int n8)
{
    const int stride = gridDim.x * 256;
    for (int i = blockIdx.x * 256 + threadIdx.x; i < n8; i += stride) {
        const float4* s = (const float4*)(src + (size_t)i * 8);
        float4 a = s[0], b = s[1];
        v8bf o;
        o[0] = (__bf16)a.x; o[1] = (__bf16)a.y; o[2] = (__bf16)a.z; o[3] = (__bf16)a.w;
        o[4] = (__bf16)b.x; o[5] = (__bf16)b.y; o[6] = (__bf16)b.z; o[7] = (__bf16)b.w;
        *(v8bf*)(dst + (size_t)i * 8) = o;
    }
}

// ---------------------------------------------------------------------------
// Router: per-token top-2 expert selection + compaction into per-expert lists.
// ---------------------------------------------------------------------------
__global__ __launch_bounds__(256) void router_kernel(
    const float* __restrict__ x,      // [T,H]
    const float* __restrict__ rw,     // [E,H]
    const float* __restrict__ rb,     // [E]
    int*         __restrict__ cnt,    // [E]
    int*         __restrict__ lists)  // [E,T]
{
    __shared__ float4 s_rw[E * H / 4];   // 32 KB
    __shared__ float  s_rb[E];

    const int tid = threadIdx.x;
    const float4* rw4 = (const float4*)rw;
    for (int i = tid; i < E * H / 4; i += 256) s_rw[i] = rw4[i];
    if (tid < E) s_rb[tid] = rb[tid];
    __syncthreads();

    const int t = blockIdx.x * 256 + tid;
    const float4* xr = (const float4*)(x + (size_t)t * H);

    float acc[E];
    #pragma unroll
    for (int e = 0; e < E; ++e) acc[e] = s_rb[e];

    for (int i = 0; i < H / 4; ++i) {
        float4 v = xr[i];
        #pragma unroll
        for (int e = 0; e < E; ++e) {
            float4 w = s_rw[e * (H / 4) + i];
            acc[e] += v.x * w.x + v.y * w.y + v.z * w.z + v.w * w.w;
        }
    }

    // top-2, ties to lower index (matches lax.top_k)
    int e0 = 0; float m0 = acc[0];
    #pragma unroll
    for (int e = 1; e < E; ++e) if (acc[e] > m0) { m0 = acc[e]; e0 = e; }
    int e1 = (e0 == 0) ? 1 : 0; float m1 = acc[e1];
    #pragma unroll
    for (int e = 0; e < E; ++e)
        if (e != e0 && e != e1 && acc[e] > m1) { m1 = acc[e]; e1 = e; }

    int p0 = atomicAdd(&cnt[e0], 1); lists[e0 * T + p0] = t;
    int p1 = atomicAdd(&cnt[e1], 1); lists[e1 * T + p1] = t;
}

// ---------------------------------------------------------------------------
// Grouped expert GEMM over bf16 inputs/weights, fp32 accumulate.
// Block = 8 wave32s, tile = 32 tokens x 256 outputs.
// Wave = 16-token sub-tile x 4 N-tiles.  Per K=64 barrier round: issue all
// 4 ds_load_b128 + 16 global_load_b128 first (distinct registers), then the
// 8 WMMAs, so load latency overlaps matrix math instead of stalling per-WMMA.
// ---------------------------------------------------------------------------
__global__ __launch_bounds__(256) void moe_gemm_kernel(
    const __bf16* __restrict__ xb,    // [T,H] bf16 activations
    const __bf16* __restrict__ Wb,    // [E,H,H] bf16 expert weights (out,in)
    const float*  __restrict__ bl,    // [E,H]
    const int*    __restrict__ cnt,   // [E]
    const int*    __restrict__ lists, // [E,T]
    float*        __restrict__ out)   // [T,H] zero-initialized accumulator
{
    const int e     = blockIdx.z;
    const int n_tok = cnt[e];
    const int t0    = blockIdx.x * 32;
    if (t0 >= n_tok) return;          // uniform across block

    __shared__ int s_tok[32];
    __shared__ __align__(16) __bf16 s_a[2][16][80];  // 32 tok x 64 k, padded rows

    const int tid = threadIdx.x;
    if (tid < 32) {
        int idx = t0 + tid;
        s_tok[tid] = (idx < n_tok) ? lists[e * T + idx] : -1;
    }
    __syncthreads();

    const int wave = tid >> 5;
    const int lane = tid & 31;
    const int ts   = wave & 1;                 // token sub-tile
    const int ng   = wave >> 1;                // N-group (64 outputs)
    const int d0   = blockIdx.y * 256 + ng * 64;
    const int ln   = lane & 15;
    const int lg   = lane >> 4;

    const __bf16* Wr[4];
    #pragma unroll
    for (int nt = 0; nt < 4; ++nt)
        Wr[nt] = Wb + (size_t)e * H * H + (size_t)(d0 + nt * 16 + ln) * H + 16 * lg;

    v8f acc[4] = {{}, {}, {}, {}};

    for (int kc = 0; kc < H; kc += 64) {
        // ---- stage A tile: 32 tok x 64 k bf16, 16B copy per thread ----
        {
            int f   = tid * 8;
            int sub = f >> 10;
            int row = (f >> 6) & 15;
            int kk  = f & 63;
            int tok = s_tok[sub * 16 + row];
            v8bf v  = {};
            if (tok >= 0) v = *(const v8bf*)(xb + (size_t)tok * H + kc + kk);
            *(v8bf*)&s_a[sub][row][kk] = v;
        }
        // prefetch next weight K-chunk toward L2/L0
        if (kc + 64 < H) {
            #pragma unroll
            for (int nt = 0; nt < 4; ++nt)
                __builtin_prefetch(Wr[nt] + kc + 64, 0, 0);
        }
        __syncthreads();

        // ---- issue ALL fragment loads for this barrier round up front ----
        union frag { v16bf v; v8bf h[2]; };
        frag a[2];
        frag b[2][4];
        #pragma unroll
        for (int u = 0; u < 2; ++u) {
            const int k0 = u * 32;
            a[u].h[0] = *(const v8bf*)&s_a[ts][ln][k0 + 16 * lg];
            a[u].h[1] = *(const v8bf*)&s_a[ts][ln][k0 + 16 * lg + 8];
            #pragma unroll
            for (int nt = 0; nt < 4; ++nt) {
                const __bf16* p = Wr[nt] + kc + k0;
                b[u][nt].h[0] = *(const v8bf*)p;
                b[u][nt].h[1] = *(const v8bf*)(p + 8);
            }
        }

        // ---- 8 WMMAs; waits become progressive instead of full stalls ----
        #pragma unroll
        for (int u = 0; u < 2; ++u) {
            #pragma unroll
            for (int nt = 0; nt < 4; ++nt) {
                acc[nt] = __builtin_amdgcn_wmma_f32_16x16x32_bf16(
                              false, a[u].v, false, b[u][nt].v,
                              (short)0, acc[nt], false, false);
            }
        }
        __syncthreads();
    }

    // ---- epilogue: exactly 2 commutative atomic adds per output element ----
    #pragma unroll
    for (int nt = 0; nt < 4; ++nt) {
        int d = d0 + nt * 16 + ln;
        float bias = bl[(size_t)e * H + d];
        #pragma unroll
        for (int v = 0; v < 8; ++v) {
            int tok = s_tok[ts * 16 + v + 8 * lg];
            if (tok >= 0)
                atomic_add_f32(out + (size_t)tok * H + d, acc[nt][v] + bias);
        }
    }
}

// ---------------------------------------------------------------------------
extern "C" void kernel_launch(void* const* d_in, const int* in_sizes, int n_in,
                              void* d_out, int out_size, void* d_ws, size_t ws_size,
                              hipStream_t stream) {
    const float* x  = (const float*)d_in[0];  // [B,S,H]
    const float* rw = (const float*)d_in[1];  // [L,E,H]
    const float* rb = (const float*)d_in[2];  // [L,E]
    const float* ew = (const float*)d_in[3];  // [L,E,H,H]
    const float* eb = (const float*)d_in[4];  // [L,E,H]
    float* out = (float*)d_out;               // [B,S,H]

    char* ws = (char*)d_ws;
    const size_t LISTS_OFF = 256;
    const size_t XB_OFF    = LISTS_OFF + sizeof(int) * (size_t)E * T;   // +512 KB
    const size_t XMID_OFF  = XB_OFF + 2ull * T * H;                     // +32 MB
    const size_t WB_OFF    = XMID_OFF + 4ull * T * H;                   // +64 MB

    int*    cnt   = (int*)ws;
    int*    lists = (int*)(ws + LISTS_OFF);
    __bf16* xb    = (__bf16*)(ws + XB_OFF);
    float*  xmid  = (float*)(ws + XMID_OFF);
    __bf16* wb    = (__bf16*)(ws + WB_OFF);                             // 32 MB

    // Convert all expert weights (both layers) to bf16 once per call.
    cvt_f32_bf16_kernel<<<2048, 256, 0, stream>>>(
        ew, wb, (int)((size_t)LAYERS * E * H * H / 8));

    for (int l = 0; l < LAYERS; ++l) {
        const float* xin  = (l == 0) ? x : xmid;
        float*       xout = (l == LAYERS - 1) ? out : xmid;

        hipMemsetAsync(cnt, 0, E * sizeof(int), stream);
        hipMemsetAsync(xout, 0, (size_t)T * H * sizeof(float), stream);

        cvt_f32_bf16_kernel<<<2048, 256, 0, stream>>>(xin, xb, T * H / 8);

        router_kernel<<<T / 256, 256, 0, stream>>>(
            xin, rw + (size_t)l * E * H, rb + (size_t)l * E, cnt, lists);

        moe_gemm_kernel<<<dim3(T / 32, H / 256, E), 256, 0, stream>>>(
            xb, wb + (size_t)l * E * H * H, eb + (size_t)l * E * H,
            cnt, lists, xout);
    }
}
